// CursorPlannerBlock_28338194219128
// MI455X (gfx1250) — matrix operand order repaired and checked by reading, not compiled, not run on hardware
//
#include <hip/hip_runtime.h>
#include <hip/hip_bf16.h>
#include <math.h>

typedef _Float16 h16;
typedef __attribute__((ext_vector_type(8)))  _Float16 v8h;
typedef __attribute__((ext_vector_type(16))) _Float16 v16h;
typedef __attribute__((ext_vector_type(8)))  float    v8f;

#define B_   8
#define S_   1024
#define T_   2048
#define D_   512
#define H_   8
#define HD_  64
#define HID_ 2048

#define BM 128
#define BN 64
#define BK 32
#define LDP 40   // padded LDS row stride (halves); 80B rows keep 16B alignment

union Frag16 { v16h v; v8h h[2]; };

__device__ __forceinline__ float gelu_tanh(float x) {
  return 0.5f * x * (1.0f + tanhf(0.7978845608028654f * (x + 0.044715f * x * x * x)));
}

__device__ __forceinline__ float block_reduce_256(float v, float* red, int isMax) {
  int t = threadIdx.x;
  red[t] = v;
  __syncthreads();
  for (int st = 128; st > 0; st >>= 1) {
    if (t < st) red[t] = isMax ? fmaxf(red[t], red[t + st]) : (red[t] + red[t + st]);
    __syncthreads();
  }
  float r = red[0];
  __syncthreads();
  return r;
}

// ---------------------------------------------------------------------------
// Generic WMMA GEMM: C = act(scale * A @ Bt^T + bias) + residual
// A: [M][K] row-major, f16 (AISF32=0) or f32 (AISF32=1, converted in staging)
// Bt: [N][K] row-major f16 (pre-transposed weights / keys)
// C: [M][N], f16 or f32. Two-level batch strides (bz = b1*bz2 + b2).
// 128x64 block tile, 4 waves x (2x4) 16x16 fragments, double-buffered LDS.
// ---------------------------------------------------------------------------
template <int AISF32>
__global__ __launch_bounds__(128)
void gemm_wmma_f16(const void* __restrict__ Av,
                   const h16* __restrict__ Bt,
                   const float* __restrict__ bias,
                   const float* __restrict__ resid,
                   void* __restrict__ Cv, int cIsF16,
                   int M, int N, int K,
                   int lda, int ldb, int ldc,
                   int bz2,
                   long long aS1, long long aS2,
                   long long bS1, long long bS2,
                   long long cS1, long long cS2,
                   float scale, int act) {
  __shared__ alignas(16) h16 As[2][BM * LDP];   // 2*128*40*2 = 20.0 KB
  __shared__ alignas(16) h16 Bs[2][BN * LDP];   // 2*64*40*2  = 10.0 KB

  const int tid  = threadIdx.x;
  const int wave = tid >> 5;          // 0..3 -> 32-row band
  const int lane = tid & 31;
  const int l16  = lane & 15;
  const int hi   = lane >> 4;
  const int m0 = blockIdx.y * BM;
  const int n0 = blockIdx.x * BN;
  const int b1 = blockIdx.z / bz2;
  const int b2 = blockIdx.z % bz2;
  const long long aOff = (long long)b1 * aS1 + (long long)b2 * aS2;
  const long long bOff = (long long)b1 * bS1 + (long long)b2 * bS2;
  const long long cOff = (long long)b1 * cS1 + (long long)b2 * cS2;

  const h16*   A16 = (const h16*)Av + aOff;
  const float* A32 = (const float*)Av + aOff;
  const h16*   Bp  = Bt + bOff;

  v8f acc[2][4];
  #pragma unroll
  for (int i = 0; i < 2; i++)
    #pragma unroll
    for (int j = 0; j < 4; j++)
      #pragma unroll
      for (int e = 0; e < 8; e++) acc[i][j][e] = 0.0f;

  const int rB  = tid >> 1;         // 0..63  (B staging row)
  const int spB = (tid & 1) * 2;    // B 16B segment pair

  auto stage = [&](int buf, int k0) {
    // ---- Bt tile: 64 rows x 32 halves ----
    const h16* bsrc = Bp + (long long)(n0 + rB) * ldb + k0;
    #pragma unroll
    for (int s = spB; s < spB + 2; s++)
      *(uint4*)&Bs[buf][rB * LDP + s * 8] = *(const uint4*)(bsrc + s * 8);
    // ---- A tile: 128 rows x 32 halves, one row per thread ----
    if constexpr (!AISF32) {
      const h16* src = A16 + (long long)(m0 + tid) * lda + k0;
      #pragma unroll
      for (int s = 0; s < 4; s++)
        *(uint4*)&As[buf][tid * LDP + s * 8] = *(const uint4*)(src + s * 8);
      if (k0 + 2 * BK < K) {                 // prefetch K+2 tile
        __builtin_prefetch(src + 2 * BK, 0, 3);
        __builtin_prefetch(bsrc + 2 * BK, 0, 3);
      }
    } else {
      const float* src = A32 + (long long)(m0 + tid) * lda + k0;
      #pragma unroll
      for (int s = 0; s < 4; s++) {
        float4 f0 = *(const float4*)(src + s * 8);
        float4 f1 = *(const float4*)(src + s * 8 + 4);
        v8h hv;
        hv[0] = (h16)f0.x; hv[1] = (h16)f0.y; hv[2] = (h16)f0.z; hv[3] = (h16)f0.w;
        hv[4] = (h16)f1.x; hv[5] = (h16)f1.y; hv[6] = (h16)f1.z; hv[7] = (h16)f1.w;
        *(v8h*)&As[buf][tid * LDP + s * 8] = hv;
      }
      if (k0 + 2 * BK < K) {
        __builtin_prefetch(src + 2 * BK, 0, 3);
        __builtin_prefetch(bsrc + 2 * BK, 0, 3);
      }
    }
  };

  auto compute = [&](int buf) {
    Frag16 fa[2], fb[4];
    #pragma unroll
    for (int mi = 0; mi < 2; mi++) {
      int ar = wave * 32 + mi * 16 + l16;
      fa[mi].h[0] = *(const v8h*)&As[buf][ar * LDP + hi * 8];       // K = hi*8 + 0..7
      fa[mi].h[1] = *(const v8h*)&As[buf][ar * LDP + 16 + hi * 8];  // K = 16+hi*8 ..
    }
    #pragma unroll
    for (int ni = 0; ni < 4; ni++) {
      int br = ni * 16 + l16;
      fb[ni].h[0] = *(const v8h*)&Bs[buf][br * LDP + hi * 8];       // K = hi*8 + 0..7
      fb[ni].h[1] = *(const v8h*)&Bs[buf][br * LDP + 16 + hi * 8];  // K = 16+hi*8 ..
    }
    #pragma unroll
    for (int mi = 0; mi < 2; mi++)
      #pragma unroll
      for (int ni = 0; ni < 4; ni++)
        acc[mi][ni] = __builtin_amdgcn_wmma_f32_16x16x32_f16(
            false, fa[mi].v, false, fb[ni].v, (short)0, acc[mi][ni], false, false);
  };

  const int nk = K / BK;
  stage(0, 0);
  __syncthreads();
  for (int kt = 0; kt < nk; kt++) {
    const int cur = kt & 1;
    if (kt + 1 < nk) stage(cur ^ 1, (kt + 1) * BK);
    compute(cur);
    __syncthreads();
  }

  h16*   C16 = (h16*)Cv;
  float* C32 = (float*)Cv;
  #pragma unroll
  for (int mi = 0; mi < 2; mi++)
    #pragma unroll
    for (int ni = 0; ni < 4; ni++) {
      int col = n0 + ni * 16 + l16;
      #pragma unroll
      for (int j = 0; j < 8; j++) {
        int row = m0 + wave * 32 + mi * 16 + hi * 8 + j;
        float v = acc[mi][ni][j] * scale;
        if (bias) v += bias[col];
        if (act == 1) v = gelu_tanh(v);
        long long idx = cOff + (long long)row * ldc + col;
        if (resid) v += resid[idx];
        if (cIsF16) C16[idx] = (h16)v; else C32[idx] = v;
      }
    }
}

// ---------------------------------------------------------------------------
// Transpose + convert to f16: out[n][k] = in[k][n]
// ---------------------------------------------------------------------------
__global__ void transpose_f16_k(const void* __restrict__ inV, int inIsF16,
                                h16* __restrict__ out,
                                int Krows, int Ncols, int ldi, int ldo,
                                int nb2,
                                long long iS1, long long iS2,
                                long long oS1, long long oS2) {
  __shared__ h16 tile[32][33];
  int b1 = blockIdx.z / nb2, b2 = blockIdx.z % nb2;
  long long iOff = (long long)b1 * iS1 + (long long)b2 * iS2;
  long long oOff = (long long)b1 * oS1 + (long long)b2 * oS2;
  int n0 = blockIdx.x * 32, k0 = blockIdx.y * 32;
  int tx = threadIdx.x, ty = threadIdx.y;  // 32 x 8
  const float* in32 = (const float*)inV;
  const h16*   in16 = (const h16*)inV;
  #pragma unroll
  for (int i = 0; i < 4; i++) {
    int k = k0 + ty + i * 8;
    float v = inIsF16 ? (float)in16[iOff + (long long)k * ldi + n0 + tx]
                      : in32[iOff + (long long)k * ldi + n0 + tx];
    tile[ty + i * 8][tx] = (h16)v;
  }
  __syncthreads();
  #pragma unroll
  for (int i = 0; i < 4; i++) {
    int n = n0 + ty + i * 8;
    out[oOff + (long long)n * ldo + k0 + tx] = tile[tx][ty + i * 8];
  }
}

// ---------------------------------------------------------------------------
// LayerNorm over last dim (optionally concat of two f32 inputs), f16 out
// ---------------------------------------------------------------------------
__global__ __launch_bounds__(256)
void ln_f16(const float* __restrict__ in1, const float* __restrict__ in2,
            int d1, int Dn, const float* __restrict__ g, const float* __restrict__ bp,
            h16* __restrict__ out) {
  __shared__ float red[256];
  long long row = blockIdx.x;
  const float* r1 = in1 + row * d1;
  const float* r2 = in2 ? in2 + row * (long long)(Dn - d1) : nullptr;
  float s = 0.0f;
  for (int c = threadIdx.x; c < Dn; c += 256) s += (c < d1) ? r1[c] : r2[c - d1];
  float mean = block_reduce_256(s, red, 0) / (float)Dn;
  float s2 = 0.0f;
  for (int c = threadIdx.x; c < Dn; c += 256) {
    float v = ((c < d1) ? r1[c] : r2[c - d1]) - mean;
    s2 += v * v;
  }
  float rstd = rsqrtf(block_reduce_256(s2, red, 0) / (float)Dn + 1e-5f);
  for (int c = threadIdx.x; c < Dn; c += 256) {
    float v = (c < d1) ? r1[c] : r2[c - d1];
    out[row * Dn + c] = (h16)((v - mean) * rstd * g[c] + bp[c]);
  }
}

// ---------------------------------------------------------------------------
// Self-attention softmax over f16 score rows, in place (probs)
// ---------------------------------------------------------------------------
__global__ __launch_bounds__(256)
void softmax_self_k(h16* __restrict__ probs, const unsigned char* __restrict__ mask) {
  __shared__ float red[256];
  long long rowIdx = blockIdx.x;                 // (b*H + h)*S + q
  h16* row = probs + rowIdx * S_;
  int b = blockIdx.x / (H_ * S_);
  const unsigned char* mrow = mask + (long long)b * S_;
  float mx = -1e30f;
  for (int c = threadIdx.x; c < S_; c += 256)
    if (!mrow[c]) mx = fmaxf(mx, (float)row[c]);
  mx = block_reduce_256(mx, red, 1);
  float s = 0.0f;
  for (int c = threadIdx.x; c < S_; c += 256)
    if (!mrow[c]) s += expf((float)row[c] - mx);
  s = block_reduce_256(s, red, 0);
  float inv = 1.0f / s;
  for (int c = threadIdx.x; c < S_; c += 256)
    row[c] = mrow[c] ? (h16)0.0f : (h16)(expf((float)row[c] - mx) * inv);
}

// ---------------------------------------------------------------------------
// Cursor two-pass softmax: expected position + gaussian-biased re-softmax
// ---------------------------------------------------------------------------
__global__ __launch_bounds__(256)
void softmax_cursor_k(float* __restrict__ logits, float* __restrict__ expPos,
                      float* __restrict__ lw, const unsigned char* __restrict__ mask) {
  __shared__ float red[256];
  long long rowIdx = blockIdx.x;                 // b*S + s
  int b = blockIdx.x / S_;
  float* row = logits + rowIdx * T_;
  float* wout = lw + rowIdx * T_;
  const unsigned char* mrow = mask + (long long)b * T_;
  // write masked logits (reference returns masked logits)
  for (int c = threadIdx.x; c < T_; c += 256)
    if (mrow[c]) row[c] = -INFINITY;
  float mx = -1e30f;
  for (int c = threadIdx.x; c < T_; c += 256)
    if (!mrow[c]) mx = fmaxf(mx, row[c]);
  mx = block_reduce_256(mx, red, 1);
  float s = 0.0f, sp = 0.0f;
  for (int c = threadIdx.x; c < T_; c += 256)
    if (!mrow[c]) { float e = expf(row[c] - mx); s += e; sp += e * (float)c; }
  s  = block_reduce_256(s, red, 0);
  sp = block_reduce_256(sp, red, 0);
  float ep = sp / s;
  if (threadIdx.x == 0) expPos[rowIdx] = ep;
  const float invDen = 1.0f / 512.0f;            // 1/(2*R^2), R=16
  float mx2 = -1e30f;
  for (int c = threadIdx.x; c < T_; c += 256)
    if (!mrow[c]) { float d = (float)c - ep; mx2 = fmaxf(mx2, row[c] - d * d * invDen); }
  mx2 = block_reduce_256(mx2, red, 1);
  float s2 = 0.0f;
  for (int c = threadIdx.x; c < T_; c += 256)
    if (!mrow[c]) { float d = (float)c - ep; s2 += expf(row[c] - d * d * invDen - mx2); }
  s2 = block_reduce_256(s2, red, 0);
  float inv = 1.0f / s2;
  for (int c = threadIdx.x; c < T_; c += 256) {
    if (mrow[c]) { wout[c] = 0.0f; continue; }
    float d = (float)c - ep;
    wout[c] = expf(row[c] - d * d * invDen - mx2) * inv;
  }
}

// gate = sigmoid(dot(hid_row, g2_w) + g2_b)
__global__ __launch_bounds__(256)
void gate_dot_k(const h16* __restrict__ hid, const float* __restrict__ w,
                const float* __restrict__ b2, float* __restrict__ gate) {
  __shared__ float red[256];
  long long row = blockIdx.x;
  const h16* hr = hid + row * HID_;
  float s = 0.0f;
  for (int c = threadIdx.x; c < HID_; c += 256) s += (float)hr[c] * w[c];
  s = block_reduce_256(s, red, 0);
  if (threadIdx.x == 0) gate[row] = 1.0f / (1.0f + expf(-(s + b2[0])));
}

__global__ void apply_gate_k(const float* __restrict__ x1, const float* __restrict__ tctxo,
                             const float* __restrict__ gate, float* __restrict__ out) {
  long long i = (long long)blockIdx.x * 256 + threadIdx.x;
  out[i] = x1[i] + gate[i / D_] * tctxo[i];
}

// ---------------------------------------------------------------------------
// Host
// ---------------------------------------------------------------------------
static void launch_gemm(hipStream_t st, const void* A, int aIsF32, const h16* Bt,
                        const float* bias, const float* resid, void* C, int cIsF16,
                        int M, int N, int K, int lda, int ldb, int ldc,
                        int bz1, int bz2,
                        long long aS1, long long aS2, long long bS1, long long bS2,
                        long long cS1, long long cS2, float scale, int act) {
  dim3 grid(N / BN, M / BM, bz1 * bz2);
  if (aIsF32)
    gemm_wmma_f16<1><<<grid, 128, 0, st>>>(A, Bt, bias, resid, C, cIsF16,
                                           M, N, K, lda, ldb, ldc, bz2,
                                           aS1, aS2, bS1, bS2, cS1, cS2, scale, act);
  else
    gemm_wmma_f16<0><<<grid, 128, 0, st>>>(A, Bt, bias, resid, C, cIsF16,
                                           M, N, K, lda, ldb, ldc, bz2,
                                           aS1, aS2, bS1, bS2, cS1, cS2, scale, act);
}

static void launch_tr(hipStream_t st, const void* in, int inIsF16, h16* out,
                      int K, int N, int ldi, int ldo, int nb1, int nb2,
                      long long iS1, long long iS2, long long oS1, long long oS2) {
  dim3 grid(N / 32, K / 32, nb1 * nb2);
  transpose_f16_k<<<grid, dim3(32, 8), 0, st>>>(in, inIsF16, out, K, N, ldi, ldo,
                                                nb2, iS1, iS2, oS1, oS2);
}

extern "C" void kernel_launch(void* const* d_in, const int* in_sizes, int n_in,
                              void* d_out, int out_size, void* d_ws, size_t ws_size,
                              hipStream_t stream) {
  const float* x    = (const float*)d_in[0];
  const unsigned char* smask = (const unsigned char*)d_in[1];
  const float* tmem = (const float*)d_in[2];
  const unsigned char* tmask = (const unsigned char*)d_in[3];
  const float* sa_g = (const float*)d_in[4];
  const float* sa_b = (const float*)d_in[5];
  const float* in_w = (const float*)d_in[6];
  const float* in_b = (const float*)d_in[7];
  const float* out_w = (const float*)d_in[8];
  const float* out_b = (const float*)d_in[9];
  const float* wq = (const float*)d_in[10];  const float* bq = (const float*)d_in[11];
  const float* wk = (const float*)d_in[12];  const float* bk = (const float*)d_in[13];
  const float* wv = (const float*)d_in[14];  const float* bv = (const float*)d_in[15];
  const float* wo = (const float*)d_in[16];  const float* bo = (const float*)d_in[17];
  const float* gn_g = (const float*)d_in[18]; const float* gn_b = (const float*)d_in[19];
  const float* g1_w = (const float*)d_in[20]; const float* g1_b = (const float*)d_in[21];
  const float* g2_w = (const float*)d_in[22]; const float* g2_b = (const float*)d_in[23];
  const float* fn_g = (const float*)d_in[24]; const float* fn_b = (const float*)d_in[25];
  const float* f1_w = (const float*)d_in[26]; const float* f1_b = (const float*)d_in[27];
  const float* f2_w = (const float*)d_in[28]; const float* f2_b = (const float*)d_in[29];

  const int BS = B_ * S_;                      // 8192
  const int BT = B_ * T_;                      // 16384
  const int D3 = 3 * D_;                       // 1536
  const int D2 = 2 * D_;                       // 1024

  char* ws = (char*)d_ws;
  size_t off = 0;
  auto alloc = [&](size_t bytes) { size_t r = off; off += (bytes + 255) & ~(size_t)255; return r; };

  h16* in_wT  = (h16*)(ws + alloc((size_t)D3 * D_ * 2));
  h16* out_wT = (h16*)(ws + alloc((size_t)D_ * D_ * 2));
  h16* wqT    = (h16*)(ws + alloc((size_t)D_ * D_ * 2));
  h16* wkT    = (h16*)(ws + alloc((size_t)D_ * D_ * 2));
  h16* wvT    = (h16*)(ws + alloc((size_t)D_ * D_ * 2));
  h16* woT    = (h16*)(ws + alloc((size_t)D_ * D_ * 2));
  h16* g1_wT  = (h16*)(ws + alloc((size_t)HID_ * D2 * 2));
  h16* f1_wT  = (h16*)(ws + alloc((size_t)HID_ * D_ * 2));
  h16* f2_wT  = (h16*)(ws + alloc((size_t)D_ * HID_ * 2));
  float* x1    = (float*)(ws + alloc((size_t)BS * D_ * 4));
  float* tctxo = (float*)(ws + alloc((size_t)BS * D_ * 4));
  float* gate  = (float*)(ws + alloc((size_t)BS * 4));
  char*  big   = ws + alloc(184549376);
  // phase A
  h16* xn    = (h16*)(big);
  h16* qkv   = (h16*)(big + 8388608);
  h16* probs = (h16*)(big + 33554432);
  h16* vT    = (h16*)(big + 167772160);
  h16* ctx   = (h16*)(big + 176160768);
  // phase B (aliases phase A, dead by then)
  h16* cq    = (h16*)(big);
  h16* ck    = (h16*)(big + 8388608);
  h16* cv    = (h16*)(big + 25165824);
  h16* cvT   = (h16*)(big + 41943040);
  h16* tctx  = (h16*)(big + 58720256);
  // phase C (aliases phase B)
  h16* gh    = (h16*)(big);
  h16* hid   = (h16*)(big + 16777216);
  h16* fh    = (h16*)(big + 50331648);

  float* outX  = (float*)d_out;
  float* outLg = outX + (size_t)BS * D_;
  float* outEp = outLg + (size_t)BS * T_;
  float* outLw = outEp + BS;

  const float scLog = 0.044194173824159216f;   // 1/sqrt(512)

  // ---- weight transposes (f32 -> f16 [N][K]) ----
  launch_tr(stream, in_w, 0, in_wT, D_, D3, D3, D_, 1, 1, 0, 0, 0, 0);
  launch_tr(stream, out_w, 0, out_wT, D_, D_, D_, D_, 1, 1, 0, 0, 0, 0);
  launch_tr(stream, wq, 0, wqT, D_, D_, D_, D_, 1, 1, 0, 0, 0, 0);
  launch_tr(stream, wk, 0, wkT, D_, D_, D_, D_, 1, 1, 0, 0, 0, 0);
  launch_tr(stream, wv, 0, wvT, D_, D_, D_, D_, 1, 1, 0, 0, 0, 0);
  launch_tr(stream, wo, 0, woT, D_, D_, D_, D_, 1, 1, 0, 0, 0, 0);
  launch_tr(stream, g1_w, 0, g1_wT, D2, HID_, HID_, D2, 1, 1, 0, 0, 0, 0);
  launch_tr(stream, f1_w, 0, f1_wT, D_, HID_, HID_, D_, 1, 1, 0, 0, 0, 0);
  launch_tr(stream, f2_w, 0, f2_wT, HID_, D_, D_, HID_, 1, 1, 0, 0, 0, 0);

  // ---- phase A: self attention ----
  ln_f16<<<BS, 256, 0, stream>>>(x, nullptr, D_, D_, sa_g, sa_b, xn);
  launch_gemm(stream, xn, 0, in_wT, in_b, nullptr, qkv, 1,
              BS, D3, D_, D_, D_, D3, 1, 1, 0, 0, 0, 0, 0, 0, 1.0f, 0);
  // vT[b][h][d][s] from v slice of qkv
  launch_tr(stream, qkv + 2 * D_, 1, vT, S_, HD_, D3, S_, B_, H_,
            (long long)S_ * D3, HD_, (long long)H_ * HD_ * S_, (long long)HD_ * S_);
  // scores: per (b,h) Q @ K^T / 8  -> probs f16
  launch_gemm(stream, qkv, 0, qkv + D_, nullptr, nullptr, probs, 1,
              S_, S_, HD_, D3, D3, S_, B_, H_,
              (long long)S_ * D3, HD_, (long long)S_ * D3, HD_,
              (long long)H_ * S_ * S_, (long long)S_ * S_, 0.125f, 0);
  softmax_self_k<<<B_ * H_ * S_, 256, 0, stream>>>(probs, smask);
  // ctx: per (b,h) probs @ V  -> ctx f16 [B,S,H,HD]
  launch_gemm(stream, probs, 0, vT, nullptr, nullptr, ctx, 1,
              S_, HD_, S_, S_, S_, D_, B_, H_,
              (long long)H_ * S_ * S_, (long long)S_ * S_,
              (long long)H_ * HD_ * S_, (long long)HD_ * S_,
              (long long)S_ * D_, HD_, 1.0f, 0);
  // x1 = x + ctx @ out_w + out_b
  launch_gemm(stream, ctx, 0, out_wT, out_b, x, x1, 0,
              BS, D_, D_, D_, D_, D_, 1, 1, 0, 0, 0, 0, 0, 0, 1.0f, 0);

  // ---- phase B: cursor cross attention ----
  launch_gemm(stream, x1, 1, wqT, bq, nullptr, cq, 1,
              BS, D_, D_, D_, D_, D_, 1, 1, 0, 0, 0, 0, 0, 0, 1.0f, 0);
  launch_gemm(stream, tmem, 1, wkT, bk, nullptr, ck, 1,
              BT, D_, D_, D_, D_, D_, 1, 1, 0, 0, 0, 0, 0, 0, 1.0f, 0);
  launch_gemm(stream, tmem, 1, wvT, bv, nullptr, cv, 1,
              BT, D_, D_, D_, D_, D_, 1, 1, 0, 0, 0, 0, 0, 0, 1.0f, 0);
  launch_tr(stream, cv, 1, cvT, T_, D_, D_, T_, B_, 1,
            (long long)T_ * D_, 0, (long long)D_ * T_, 0);
  // logits: per b, cq @ ck^T / sqrt(D) -> d_out
  launch_gemm(stream, cq, 0, ck, nullptr, nullptr, outLg, 0,
              S_, T_, D_, D_, D_, T_, B_, 1,
              (long long)S_ * D_, 0, (long long)T_ * D_, 0,
              (long long)S_ * T_, 0, scLog, 0);
  softmax_cursor_k<<<BS, 256, 0, stream>>>(outLg, outEp, outLw, tmask);
  // text_ctx: per b, local_weights(f32) @ cv -> tctx f16
  launch_gemm(stream, outLw, 1, cvT, nullptr, nullptr, tctx, 1,
              S_, D_, T_, T_, T_, D_, B_, 1,
              (long long)S_ * T_, 0, (long long)D_ * T_, 0,
              (long long)S_ * D_, 0, 1.0f, 0);
  launch_gemm(stream, tctx, 0, woT, bo, nullptr, tctxo, 0,
              BS, D_, D_, D_, D_, D_, 1, 1, 0, 0, 0, 0, 0, 0, 1.0f, 0);

  // ---- phase C: gate + FFN ----
  ln_f16<<<BS, 256, 0, stream>>>(x1, tctxo, D_, D2, gn_g, gn_b, gh);
  launch_gemm(stream, gh, 0, g1_wT, g1_b, nullptr, hid, 1,
              BS, HID_, D2, D2, D2, HID_, 1, 1, 0, 0, 0, 0, 0, 0, 1.0f, 1);
  gate_dot_k<<<BS, 256, 0, stream>>>(hid, g2_w, g2_b, gate);
  apply_gate_k<<<(BS * D_) / 256, 256, 0, stream>>>(x1, tctxo, gate, outX);
  ln_f16<<<BS, 256, 0, stream>>>(outX, nullptr, D_, D_, fn_g, fn_b, fh);
  launch_gemm(stream, fh, 0, f1_wT, f1_b, nullptr, hid, 1,
              BS, HID_, D_, D_, D_, HID_, 1, 1, 0, 0, 0, 0, 0, 0, 1.0f, 1);
  launch_gemm(stream, hid, 0, f2_wT, f2_b, outX, outX, 0,
              BS, D_, HID_, HID_, HID_, D_, 1, 1, 0, 0, 0, 0, 0, 0, 1.0f, 0);

  (void)in_sizes; (void)n_in; (void)out_size; (void)ws_size;
}
